// Router_21457656610885
// MI455X (gfx1250) — matrix operand order repaired and checked by reading, not compiled, not run on hardware
//
#include <hip/hip_runtime.h>
#include <hip/hip_fp16.h>
#include <math.h>

typedef __attribute__((ext_vector_type(16))) _Float16 v16h;
typedef __attribute__((ext_vector_type(8)))  _Float16 v8h;
typedef __attribute__((ext_vector_type(8)))  float    v8f;
typedef __attribute__((ext_vector_type(4)))  float    v4f;

#define N_TOKENS 131072
#define HIDDEN   1024
#define CLUSTER  64
#define EXPERTS  64

#define BLOCK_TOKENS 256   // 8 waves x 2 M-tiles x 16 tokens
#define WPAD 1032          // HIDDEN + 8 halves pad (bank-conflict mitigation)
#define CPAD 72            // 64 + 8 halves pad
#define DPAD 72

// LDS layout (bytes)
#define OFF_WT   0          // 64 * WPAD * 2 = 132096; DEAD after GEMM1 -> reused as
                            //   fp32 accumulator parking: 64 tiles * 32 lanes * 48 B = 98304
#define ACC_LANE_STRIDE 48  // 32B payload + 16B pad (bank spreading)
#define ACC_TILE_STRIDE (32 * ACC_LANE_STRIDE)   // 1536
#define OFF_CN   132096     // 64 * CPAD * 2 = 9216
#define OFF_CSQ  141312     // 64 * 4 = 256
#define OFF_D    141568     // 8 waves * 16*DPAD*2 = 18432 (g staging, reused across M-tiles)
#define LDS_TOTAL 160000

__device__ __forceinline__ v16h cat8h(v8h lo, v8h hi) {
    return __builtin_shufflevector(lo, hi, 0,1,2,3,4,5,6,7,8,9,10,11,12,13,14,15);
}

__global__ __launch_bounds__(256)
__attribute__((amdgpu_waves_per_eu(4)))
void router_topk_kernel(const float* __restrict__ x,
                        const float* __restrict__ W,
                        const float* __restrict__ bias,
                        const float* __restrict__ centroids,
                        float* __restrict__ out)
{
    extern __shared__ char smem[];
    _Float16* wt  = (_Float16*)(smem + OFF_WT);    // [64][WPAD]  W transposed, f16
    _Float16* cn  = (_Float16*)(smem + OFF_CN);    // [64][CPAD]  normalized centroids, f16
    float*    csq = (float*)   (smem + OFF_CSQ);   // [64]
    _Float16* dls = (_Float16*)(smem + OFF_D);     // per-wave [16][DPAD] g staging

    const int tid  = threadIdx.x;
    const int wave = tid >> 5;
    const int lane = tid & 31;
    const int r0   = lane & 15;
    const int half = lane >> 4;

    // ---- Stage W into LDS, transposed, f16 (vectorized b128 reads) ----
    {
        const v4f* W4 = (const v4f*)W;
        #pragma unroll 4
        for (int j = 0; j < (HIDDEN * CLUSTER) / (256 * 4); ++j) {
            int i4 = tid + 256 * j;
            int k = i4 >> 4;                  // W row (k)
            int n = (i4 & 15) * 4;            // 4 consecutive columns
            v4f v = W4[i4];
            wt[(n + 0) * WPAD + k] = (_Float16)v.x;
            wt[(n + 1) * WPAD + k] = (_Float16)v.y;
            wt[(n + 2) * WPAD + k] = (_Float16)v.z;
            wt[(n + 3) * WPAD + k] = (_Float16)v.w;
        }
    }

    // ---- Normalize centroids into LDS (two-pass, low VGPR) ----
    if (tid < EXPERTS) {
        float s = 0.f;
        for (int c = 0; c < CLUSTER; ++c) { float v = centroids[tid * CLUSTER + c]; s += v * v; }
        float inv = 1.0f / fmaxf(sqrtf(s), 1e-8f);
        float s2 = 0.f;
        for (int c = 0; c < CLUSTER; ++c) {
            float nv = centroids[tid * CLUSTER + c] * inv;
            s2 += nv * nv;
            cn[tid * CPAD + c] = (_Float16)nv;
        }
        csq[tid] = s2;
    }
    __syncthreads();

    const int waveBase = blockIdx.x * BLOCK_TOKENS + wave * 32;   // 2 M-tiles of 16
    const float* xrow0 = x + (size_t)(waveBase + r0) * HIDDEN;
    const float* xrow1 = x + (size_t)(waveBase + 16 + r0) * HIDDEN;

    // ---- GEMM1: h = x @ W  (two M-tiles share every B fragment) ----
    v8f acc[2][4];
    #pragma unroll
    for (int u = 0; u < 2; ++u)
        #pragma unroll
        for (int t = 0; t < 4; ++t)
            #pragma unroll
            for (int r = 0; r < 8; ++r) acc[u][t][r] = 0.f;

    for (int kk = 0; kk < HIDDEN; kk += 32) {
        v16h a[2];
        #pragma unroll
        for (int u = 0; u < 2; ++u) {
            const float* base = (u == 0 ? xrow0 : xrow1) + kk;
            v4f f0 = __builtin_nontemporal_load((const v4f*)(base + 8 * half));
            v4f f1 = __builtin_nontemporal_load((const v4f*)(base + 8 * half) + 1);
            v4f f2 = __builtin_nontemporal_load((const v4f*)(base + 16 + 8 * half));
            v4f f3 = __builtin_nontemporal_load((const v4f*)(base + 16 + 8 * half) + 1);
            v16h av;
            av[0]=(_Float16)f0.x;  av[1]=(_Float16)f0.y;  av[2]=(_Float16)f0.z;  av[3]=(_Float16)f0.w;
            av[4]=(_Float16)f1.x;  av[5]=(_Float16)f1.y;  av[6]=(_Float16)f1.z;  av[7]=(_Float16)f1.w;
            av[8]=(_Float16)f2.x;  av[9]=(_Float16)f2.y;  av[10]=(_Float16)f2.z; av[11]=(_Float16)f2.w;
            av[12]=(_Float16)f3.x; av[13]=(_Float16)f3.y; av[14]=(_Float16)f3.z; av[15]=(_Float16)f3.w;
            a[u] = av;
        }
        #pragma unroll
        for (int t = 0; t < 4; ++t) {
            int n = r0 + 16 * t;
            const _Float16* bp = &wt[n * WPAD + kk + 8 * half];   // B: table K order (8h.., 16+8h..)
            v8h lo = *(const v8h*)bp;
            v8h hi = *(const v8h*)(bp + 16);
            v16h bf = cat8h(lo, hi);
            acc[0][t] = __builtin_amdgcn_wmma_f32_16x16x32_f16(
                false, a[0], false, bf, (short)0, acc[0][t], false, false);
            acc[1][t] = __builtin_amdgcn_wmma_f32_16x16x32_f16(
                false, a[1], false, bf, (short)0, acc[1][t], false, false);
        }
    }

    // wt region is dead once every wave has passed its k-loop: park the raw fp32
    // accumulators there. This frees all 64 acc VGPRs and lets the whole epilogue
    // run as ONE rolled loop (single static copy of erf/GEMM2/top-2).
    __syncthreads();

    char* accB = smem;   // fp32 acc parking (old wt region)
    #pragma unroll
    for (int u = 0; u < 2; ++u)
        #pragma unroll
        for (int t = 0; t < 4; ++t) {
            int T = (wave * 2 + u) * 4 + t;
            char* p = accB + T * ACC_TILE_STRIDE + lane * ACC_LANE_STRIDE;
            *(v4f*)(p)      = __builtin_shufflevector(acc[u][t], acc[u][t], 0, 1, 2, 3);
            *(v4f*)(p + 16) = __builtin_shufflevector(acc[u][t], acc[u][t], 4, 5, 6, 7);
        }

    _Float16* dw = dls + wave * 16 * DPAD;

    // ---- Epilogue per M-tile, fully rolled (same-wave DS ordering throughout) ----
    #pragma unroll 1
    for (int u = 0; u < 2; ++u) {
        const int tokenBase = waveBase + 16 * u;

        // Bias + exact GELU from parked acc; stage UNNORMALIZED g (f16).
        // Normalization folds into GEMM2: d.c = invn[row] * (g.c).
        float s[8];
        #pragma unroll
        for (int r = 0; r < 8; ++r) s[r] = 0.f;
        #pragma unroll 1
        for (int t = 0; t < 4; ++t) {
            int T = (wave * 2 + u) * 4 + t;
            const char* p = accB + T * ACC_TILE_STRIDE + lane * ACC_LANE_STRIDE;
            v4f lo = *(const v4f*)(p);
            v4f hi = *(const v4f*)(p + 16);
            float bb = bias[r0 + 16 * t];
            #pragma unroll
            for (int r = 0; r < 8; ++r) {
                float v = ((r < 4) ? lo[r] : hi[r - 4]) + bb;
                float g = 0.5f * v * (1.0f + erff(v * 0.70710678118654752440f));
                s[r] += g * g;
                dw[(r + 8 * half) * DPAD + r0 + 16 * t] = (_Float16)g;
            }
        }
        #pragma unroll
        for (int m = 1; m <= 8; m <<= 1)
            #pragma unroll
            for (int r = 0; r < 8; ++r) s[r] += __shfl_xor(s[r], m, 32);

        float invn[8], dsq[8];
        #pragma unroll
        for (int r = 0; r < 8; ++r) {
            float iv = 1.0f / fmaxf(sqrtf(s[r]), 1e-8f);
            invn[r] = iv;
            dsq[r] = s[r] * iv * iv;
        }

        // GEMM2: scores = g @ cN^T
        v8f sc[4];
        #pragma unroll
        for (int t = 0; t < 4; ++t)
            #pragma unroll
            for (int r = 0; r < 8; ++r) sc[t][r] = 0.f;
        #pragma unroll
        for (int kc = 0; kc < CLUSTER; kc += 32) {
            v8h alo = *(const v8h*)&dw[r0 * DPAD + kc + 8 * half];
            v8h ahi = *(const v8h*)&dw[r0 * DPAD + kc + 16 + 8 * half];
            v16h a2 = cat8h(alo, ahi);
            #pragma unroll
            for (int t = 0; t < 4; ++t) {
                int e = r0 + 16 * t;
                const _Float16* bp = &cn[e * CPAD + kc + 8 * half];
                v8h blo = *(const v8h*)bp;
                v8h bhi = *(const v8h*)(bp + 16);
                v16h b2 = cat8h(blo, bhi);
                sc[t] = __builtin_amdgcn_wmma_f32_16x16x32_f16(
                    false, a2, false, b2, (short)0, sc[t], false, false);
            }
        }

        // Single pass: dist -> exp-sum + two-smallest tracking.
        // dist <= ~2 (unit vectors) so exp(-dist) needs no max-shift;
        // softmax is monotone decreasing in dist -> 2 smallest dists = top-2 probs.
        float d1[8], d2[8], ssum[8]; int i1[8], i2[8];
        #pragma unroll
        for (int r = 0; r < 8; ++r) { d1[r] = 1e30f; d2[r] = 1e30f; ssum[r] = 0.f; i1[r] = 0; i2[r] = 0; }
        #pragma unroll
        for (int t = 0; t < 4; ++t) {
            float cqt = csq[r0 + 16 * t];
            int e = r0 + 16 * t;
            #pragma unroll
            for (int r = 0; r < 8; ++r) {
                float sq = dsq[r] + cqt - 2.0f * invn[r] * sc[t][r];
                float dist = sqrtf(fmaxf(sq, 0.0f));
                ssum[r] += __expf(-dist);
                if (dist < d1[r]) { d2[r] = d1[r]; i2[r] = i1[r]; d1[r] = dist; i1[r] = e; }
                else if (dist < d2[r]) { d2[r] = dist; i2[r] = e; }
            }
        }
        #pragma unroll
        for (int m = 1; m <= 8; m <<= 1) {
            #pragma unroll
            for (int r = 0; r < 8; ++r) {
                ssum[r] += __shfl_xor(ssum[r], m, 32);
                float q1 = __shfl_xor(d1[r], m, 32); int j1 = __shfl_xor(i1[r], m, 32);
                float q2 = __shfl_xor(d2[r], m, 32); int j2 = __shfl_xor(i2[r], m, 32);
                if (q1 < d1[r]) {
                    float nd = (q2 < d1[r]) ? q2 : d1[r];
                    int   nj = (q2 < d1[r]) ? j2 : i1[r];
                    d1[r] = q1; i1[r] = j1; d2[r] = nd; i2[r] = nj;
                } else if (q1 < d2[r]) {
                    d2[r] = q1; i2[r] = j1;
                }
            }
        }

        // Output: probs [0, 2N), indices-as-float [2N, 4N); non-temporal (write-once)
        if (r0 == 0) {
            #pragma unroll
            for (int r = 0; r < 8; ++r) {
                float rs = 1.0f / ssum[r];
                int token = tokenBase + r + 8 * half;
                __builtin_nontemporal_store(__expf(-d1[r]) * rs, out + (size_t)token * 2 + 0);
                __builtin_nontemporal_store(__expf(-d2[r]) * rs, out + (size_t)token * 2 + 1);
                __builtin_nontemporal_store((float)i1[r],
                    out + (size_t)N_TOKENS * 2 + (size_t)token * 2 + 0);
                __builtin_nontemporal_store((float)i2[r],
                    out + (size_t)N_TOKENS * 2 + (size_t)token * 2 + 1);
            }
        }
    }
}

extern "C" void kernel_launch(void* const* d_in, const int* in_sizes, int n_in,
                              void* d_out, int out_size, void* d_ws, size_t ws_size,
                              hipStream_t stream) {
    (void)in_sizes; (void)n_in; (void)d_ws; (void)ws_size; (void)out_size;
    const float* x = (const float*)d_in[0];
    const float* W = (const float*)d_in[1];
    const float* b = (const float*)d_in[2];
    const float* c = (const float*)d_in[3];
    // d_in[4] is k == 2 (hardcoded)

    (void)hipFuncSetAttribute(reinterpret_cast<const void*>(router_topk_kernel),
                              hipFuncAttributeMaxDynamicSharedMemorySize, LDS_TOTAL);

    dim3 grid(N_TOKENS / BLOCK_TOKENS), block(256);
    router_topk_kernel<<<grid, block, LDS_TOTAL, stream>>>(x, W, b, c, (float*)d_out);
}
